// SGCModel_61538291417128
// MI455X (gfx1250) — matrix operand order repaired and checked by reading, not compiled, not run on hardware
//
#include <hip/hip_runtime.h>

typedef __attribute__((ext_vector_type(16))) _Float16 v16h;
typedef __attribute__((ext_vector_type(8)))  float    v8f;

// ---------------------------------------------------------------------------
// Step 1: degree (self-loop folded in as init value 1.0)
// ---------------------------------------------------------------------------
__global__ void init_deg_kernel(float* __restrict__ deg, int n) {
    int i = blockIdx.x * blockDim.x + threadIdx.x;
    if (i < n) deg[i] = 1.0f;   // self-loop contributes 1 to every node's degree
}

__global__ void degree_kernel(const long long* __restrict__ dst,
                              float* __restrict__ deg, int e) {
    int i = blockIdx.x * blockDim.x + threadIdx.x;
    if (i < e) atomicAdd(&deg[(int)dst[i]], 1.0f);
}

__global__ void dinv_kernel(float* __restrict__ deg, int n) {
    int i = blockIdx.x * blockDim.x + threadIdx.x;
    if (i < n) deg[i] = rsqrtf(deg[i]);   // deg >= 1 always, no zero guard needed
}

// ---------------------------------------------------------------------------
// Step 2: seed propagation buffer with the self-loop term: out = dinv^2 * feat
// (avoids atomics for the N self-loop messages). One float2 per thread.
// ---------------------------------------------------------------------------
__global__ void seed_kernel(const float* __restrict__ feat,
                            const float* __restrict__ dinv,
                            float* __restrict__ out, int n) {
    int t = blockIdx.x * blockDim.x + threadIdx.x;   // over n*32
    if (t >= n * 32) return;
    int node = t >> 5;
    float di = dinv[node];
    float s  = di * di;
    const float2* fv = (const float2*)feat;
    float2 v = fv[t];
    float2* ov = (float2*)out;
    ov[t] = make_float2(v.x * s, v.y * s);
}

// ---------------------------------------------------------------------------
// Step 3: edge scatter: out[dst] += dinv[src]*dinv[dst] * feat[src]
// One wave (32 lanes) per edge, float2 per lane -> fully coalesced 256B gather
// per edge, f32 atomics into L2 (feature matrix + dest fit in 192MB L2).
// ---------------------------------------------------------------------------
__global__ void scatter_kernel(const float* __restrict__ feat,
                               const long long* __restrict__ src,
                               const long long* __restrict__ dst,
                               const float* __restrict__ dinv,
                               float* __restrict__ out, int e) {
    long long t = (long long)blockIdx.x * blockDim.x + threadIdx.x;
    if (t >= (long long)e * 32) return;
    int eidx = (int)(t >> 5);
    int lane = (int)(t & 31);
    int s = (int)src[eidx];
    int d = (int)dst[eidx];
    float norm = dinv[s] * dinv[d];
    const float2* fv = (const float2*)feat;
    float2 v = fv[(size_t)s * 32 + lane];
    atomicAdd(&out[(size_t)d * 64 + lane * 2 + 0], v.x * norm);
    atomicAdd(&out[(size_t)d * 64 + lane * 2 + 1], v.y * norm);
}

// ---------------------------------------------------------------------------
// Step 4: fused GEMM + bias (+ ReLU):  out[n, OUTC] = act(A[n,64] @ W^T + b)
// W is [OUTC, 64] row-major, so B[k][o] = W[o][k].
// Block = 256 threads = 8 waves; each wave owns one 16-row tile (128 rows/blk).
// W and the A tile staged in LDS as f16; v_wmma_f32_16x16x32_f16, K=64 as two
// K=32 chunks, f32 accumulate seeded with the bias.
// Bounds checks are hoisted into a single wave-uniform branch (readfirstlane)
// so the 781 full tiles run a guard-free, exec-mask-free path.
// Fragment layouts used here (wave32; A and B share a contiguous k order):
//   A 16x32 f16 : lane l holds row M=l&15; VGPR v (halves 2v,2v+1) holds
//                 K = 32*kc + 16*(l>=16) + 2*v + {0,1}
//   B 32x16 f16 : lane l holds col N=l&15; half i holds K = 32*kc + i + 16*(l>=16)
//   C/D 16x16 f32: VGPR r -> row M = r + 8*(l>=16), col N = l&15
// ---------------------------------------------------------------------------
template <int OUTC, bool RELU>
__global__ __launch_bounds__(256)
void gemm_bias_act_wmma(const float* __restrict__ A,
                        const float* __restrict__ W,
                        const float* __restrict__ bias,
                        float* __restrict__ out, int n) {
    __shared__ _Float16 sW[OUTC * 64] __attribute__((aligned(32)));
    __shared__ _Float16 sA[8][16 * 64] __attribute__((aligned(32)));

    const int tid  = threadIdx.x;
    const int wave = tid >> 5;
    const int lane = tid & 31;

    // Stage weights (f32 -> f16), coalesced.
    for (int i = tid; i < OUTC * 64; i += 256)
        sW[i] = (_Float16)W[i];

    // Wave-uniform tile base + full-tile predicate (scalar branch, no exec ops).
    const int row0 = __builtin_amdgcn_readfirstlane(blockIdx.x * 128 + wave * 16);
    const bool full = (row0 + 16 <= n);

    // Stage this wave's 16x64 A tile (f32 -> f16), coalesced.
    const float* __restrict__ Atile = A + (size_t)row0 * 64;
    if (full) {
        for (int j = 0; j < 32; ++j) {
            int idx = lane + 32 * j;                  // 0..1023 inside 16x64 tile
            sA[wave][idx] = (_Float16)Atile[idx];
        }
    } else {
        for (int j = 0; j < 32; ++j) {
            int idx = lane + 32 * j;
            int r = idx >> 6;
            float v = (row0 + r < n) ? Atile[idx] : 0.0f;
            sA[wave][idx] = (_Float16)v;
        }
    }
    __syncthreads();

    const int mrow = lane & 15;
    const int hi   = lane >> 4;           // 0: lanes 0-15, 1: lanes 16-31

    // Build both K-chunk A fragments once; reuse across all column tiles.
    union AF { v16h h; unsigned int u[8]; };
    AF afrag[2];
    for (int kc = 0; kc < 2; ++kc) {
        for (int v = 0; v < 8; ++v) {
            int K = kc * 32 + hi * 16 + v * 2;
            afrag[kc].u[v] = *(const unsigned int*)&sA[wave][mrow * 64 + K];
        }
    }

    for (int ct = 0; ct < OUTC / 16; ++ct) {
        const int col = ct * 16 + mrow;
        const float bv = bias[col];
        v8f acc;
        for (int r = 0; r < 8; ++r) acc[r] = bv;

        for (int kc = 0; kc < 2; ++kc) {
            // 16 contiguous halves, 32B-aligned -> single b128-class LDS load.
            const v16h b = *(const v16h*)&sW[col * 64 + kc * 32 + hi * 16];
            acc = __builtin_amdgcn_wmma_f32_16x16x32_f16(
                false, afrag[kc].h, false, b, (short)0, acc, false, false);
        }

        // Store: lane writes rows (hi*8 .. hi*8+7) at fixed column `col`;
        // lanes 0..15 cover 16 consecutive columns -> 64B contiguous per row.
        float* __restrict__ obase = out + (size_t)(row0 + hi * 8) * OUTC + col;
        if (full) {
            for (int r = 0; r < 8; ++r) {
                float v = acc[r];
                if (RELU) v = fmaxf(v, 0.0f);
                obase[(size_t)r * OUTC] = v;
            }
        } else {
            for (int r = 0; r < 8; ++r) {
                if (row0 + hi * 8 + r < n) {
                    float v = acc[r];
                    if (RELU) v = fmaxf(v, 0.0f);
                    obase[(size_t)r * OUTC] = v;
                }
            }
        }
    }
}

// ---------------------------------------------------------------------------
// Host launch
// ---------------------------------------------------------------------------
extern "C" void kernel_launch(void* const* d_in, const int* in_sizes, int n_in,
                              void* d_out, int out_size, void* d_ws, size_t ws_size,
                              hipStream_t stream) {
    const float*     x  = (const float*)d_in[0];        // [n, 64]
    const long long* ei = (const long long*)d_in[1];    // [2, e] int64
    const float*     W1 = (const float*)d_in[2];        // [64, 64]
    const float*     b1 = (const float*)d_in[3];        // [64]
    const float*     W2 = (const float*)d_in[4];        // [64, 64]
    const float*     b2 = (const float*)d_in[5];        // [64]
    const float*     W3 = (const float*)d_in[6];        // [16, 64]
    const float*     b3 = (const float*)d_in[7];        // [16]
    float* out = (float*)d_out;                          // [n, 16]

    const int n = in_sizes[0] / 64;
    const int e = in_sizes[1] / 2;
    const long long* src = ei;
    const long long* dst = ei + e;

    // Workspace: dinv [n] | bufA [n*64] | bufB [n*64]  (~52 MB)
    float* dinv = (float*)d_ws;
    float* bufA = dinv + n;
    float* bufB = bufA + (size_t)n * 64;

    const int T = 256;
    const int gN    = (n + T - 1) / T;
    const int gE    = (e + T - 1) / T;
    const int gSeed = (n * 32 + T - 1) / T;
    const int gScat = (int)(((long long)e * 32 + T - 1) / T);
    const int gGemm = (n + 127) / 128;

    // deg / dinv (shared by both propagation rounds)
    init_deg_kernel<<<gN, T, 0, stream>>>(dinv, n);
    degree_kernel<<<gE, T, 0, stream>>>(dst, dinv, e);
    dinv_kernel<<<gN, T, 0, stream>>>(dinv, n);

    // Round 1: prop(x) -> bufA; h1 = relu(bufA @ W1^T + b1) -> bufB
    seed_kernel<<<gSeed, T, 0, stream>>>(x, dinv, bufA, n);
    scatter_kernel<<<gScat, T, 0, stream>>>(x, src, dst, dinv, bufA, e);
    gemm_bias_act_wmma<64, true><<<gGemm, T, 0, stream>>>(bufA, W1, b1, bufB, n);

    // Round 2: prop(h1) -> bufA; h2 = relu(bufA @ W2^T + b2) -> bufB
    seed_kernel<<<gSeed, T, 0, stream>>>(bufB, dinv, bufA, n);
    scatter_kernel<<<gScat, T, 0, stream>>>(bufB, src, dst, dinv, bufA, e);
    gemm_bias_act_wmma<64, true><<<gGemm, T, 0, stream>>>(bufA, W2, b2, bufB, n);

    // Final linear: out = h2 @ W3^T + b3
    gemm_bias_act_wmma<16, false><<<gGemm, T, 0, stream>>>(bufB, W3, b3, out, n);
}
